// SDK_ESN_Infer_63488206569596
// MI455X (gfx1250) — 8-bit matrix operand order repaired and checked by reading, compiled, not run on hardware
//
#include <hip/hip_runtime.h>
#include <math.h>

typedef __attribute__((ext_vector_type(8))) int v8i;
typedef unsigned __attribute__((ext_vector_type(4))) u32x4;
typedef unsigned __attribute__((ext_vector_type(8))) u32x8;

#define Bz 32
#define Tz 512
#define Fz 128
#define Hz 2048
#define Vz 32
#define NBLK 32
#define NTHR 256

#define WR_OFF   0u
#define WI_OFF   (WR_OFF + (Hz/16)*(Hz/64)*1024u)
#define HCUR_OFF (WI_OFF + (Hz/16)*(Fz/64)*1024u)
#define HQ_OFF   (HCUR_OFF + Bz*Hz*4u)
#define UQ_OFF   (HQ_OFF + (Bz/16)*(Hz/64)*1024u)
#define BAR_OFF  (UQ_OFF + (Bz/16)*(Fz/64)*1024u)

#define LDSWR_OFF 0u
#define LDSWI_OFF 131072u
#define LDSA_OFF  139264u
#define LDSU_OFF  204800u
#define LDS_TOTAL 208896u

__global__ void pack_res_kernel(const int* __restrict__ W, char* __restrict__ out) {
    int o = blockIdx.x * blockDim.x + threadIdx.x;
    int tile = o >> 10;
    int nt = tile >> 5;
    int kt = tile & 31;
    int within = o & 1023;
    int lane = within >> 5;
    int v = (within >> 2) & 7;
    int c = within & 3;
    int half = lane >> 4;
    int n = lane & 15;
    int k = ((v >> 2) << 5) | (half << 4) | ((v & 3) << 2) | c;
    out[o] = (char)W[(nt * 16 + n) * Hz + kt * 64 + k];
}

__global__ void pack_in_kernel(const int* __restrict__ W, char* __restrict__ out) {
    int o = blockIdx.x * blockDim.x + threadIdx.x;
    int tile = o >> 10;
    int nt = tile >> 1;
    int kt = tile & 1;
    int within = o & 1023;
    int lane = within >> 5;
    int v = (within >> 2) & 7;
    int c = within & 3;
    int half = lane >> 4;
    int n = lane & 15;
    int k = ((v >> 2) << 5) | (half << 4) | ((v & 3) << 2) | c;
    out[o] = (char)W[(nt * 16 + n) * Fz + kt * 64 + k];
}

__global__ void init_kernel(float* __restrict__ h_cur, unsigned* __restrict__ bar) {
    int g = blockIdx.x * blockDim.x + threadIdx.x;
    if (g < Bz * Hz) h_cur[g] = 0.0f;
    if (g < 2) bar[g] = 0u;
}

__device__ __forceinline__ void tdm_load_2d(unsigned lds_off, const void* gaddr,
                                            unsigned rowBytes, unsigned rows) {
    unsigned long long ga = (unsigned long long)gaddr;
    u32x4 g0;
    g0[0] = 1u;
    g0[1] = lds_off;
    g0[2] = (unsigned)(ga & 0xFFFFFFFFu);
    g0[3] = (unsigned)((ga >> 32) & 0x01FFFFFFu)
            | 0x80000000u;
    u32x8 g1;
    g1[0] = 0u;
    g1[1] = (rowBytes & 0xFFFFu) << 16;
    g1[2] = ((rowBytes >> 16) & 0xFFFFu) | ((rows & 0xFFFFu) << 16);
    g1[3] = ((rows >> 16) & 0xFFFFu) | ((rowBytes & 0xFFFFu) << 16);
    g1[4] = rows & 0xFFFFu;
    g1[5] = rowBytes;
    g1[6] = 0u;
    g1[7] = 0u;
    asm volatile("tensor_load_to_lds %0, %1, null, null"
                 :: "s"(g0), "s"(g1) : "memory");
}

__device__ __forceinline__ void async_b128(unsigned lds_off, const char* gsrc) {
    asm volatile("global_load_async_to_lds_b128 %0, %1, off"
                 :: "v"(lds_off), "v"(gsrc) : "memory");
}
__device__ __forceinline__ void wait_async0() {
    asm volatile("s_wait_asynccnt 0x0" ::: "memory");
}

__device__ __forceinline__ void grid_barrier(unsigned* bar) {
    __syncthreads();
    if (threadIdx.x == 0) {
        unsigned gen = __hip_atomic_load(&bar[1], __ATOMIC_ACQUIRE, __HIP_MEMORY_SCOPE_AGENT);
        unsigned arr = __hip_atomic_fetch_add(&bar[0], 1u, __ATOMIC_ACQ_REL, __HIP_MEMORY_SCOPE_AGENT);
        if (arr == NBLK - 1) {
            __hip_atomic_store(&bar[0], 0u, __ATOMIC_RELAXED, __HIP_MEMORY_SCOPE_AGENT);
            __hip_atomic_fetch_add(&bar[1], 1u, __ATOMIC_RELEASE, __HIP_MEMORY_SCOPE_AGENT);
        } else {
            while (__hip_atomic_load(&bar[1], __ATOMIC_ACQUIRE, __HIP_MEMORY_SCOPE_AGENT) == gen)
                __builtin_amdgcn_s_sleep(1);
        }
    }
    __syncthreads();
}

__device__ __forceinline__ int a_pack_addr(int b, int k, int kTiles) {
    int kt = k >> 6, kk = k & 63;
    int vv = ((kk >> 5) << 2) | ((kk >> 2) & 3);
    int hf = (kk >> 4) & 1;
    int c = kk & 3;
    int mt = b >> 4, lane_lo = b & 15;
    return ((mt * kTiles + kt) << 10) | ((hf * 16 + lane_lo) << 5) | (vv << 2) | c;
}

__device__ __forceinline__ void project_row(const float* __restrict__ h_cur,
                                            const float* __restrict__ Wout,
                                            const float* __restrict__ bout,
                                            float* __restrict__ out,
                                            int wave, int lane, int t) {
    #pragma unroll
    for (int p = 0; p < 4; ++p) {
        int pair = wave * 4 + p;
        int b = pair >> 5;
        int v = pair & 31;
        float acc = 0.0f;
        for (int i = lane; i < Hz; i += 32)
            acc += h_cur[b * Hz + i] * Wout[v * Hz + i];
        #pragma unroll
        for (int off = 16; off > 0; off >>= 1)
            acc += __shfl_down(acc, off, 32);
        if (lane == 0)
            out[(b * Tz + t) * Vz + v] = acc + bout[v];
    }
}

__global__ void __launch_bounds__(NTHR, 1) esn_main_kernel(
    const float* __restrict__ x, const float* __restrict__ S_res,
    const float* __restrict__ S_in, const float* __restrict__ Wout,
    const float* __restrict__ bout, const char* __restrict__ Wr_p,
    const char* __restrict__ Wi_p, float* __restrict__ h_cur,
    char* __restrict__ h_qp, char* __restrict__ u_qp,
    unsigned* __restrict__ bar, float* __restrict__ out) {
    extern __shared__ char lds_base[];

    const int tid = threadIdx.x;
    const int gtid = blockIdx.x * NTHR + tid;
    const int lane = tid & 31;
    const int wave = gtid >> 5;
    const int wl = tid >> 5;
    const int mt = wave & 1;
    const int ntl = wl >> 1;
    const int nt = blockIdx.x * 4 + ntl;
    const int half = lane >> 4;
    const int i_col = nt * 16 + (lane & 15);

    const float sr = S_res[i_col] * (1.0f / 49.0f);
    const float si = S_in[i_col] * (1.0f / 49.0f);

    if (wl == 0) {
        tdm_load_2d(LDSWR_OFF, Wr_p + (size_t)blockIdx.x * 4 * (Hz / 64) * 1024,
                    1024u, 4 * (Hz / 64));
        tdm_load_2d(LDSWI_OFF, Wi_p + (size_t)blockIdx.x * 4 * (Fz / 64) * 1024,
                    1024u, 4 * (Fz / 64));
        __builtin_amdgcn_s_wait_tensorcnt(0);
    }
    __syncthreads();

    const char* ldsA  = lds_base + LDSA_OFF  + ((mt * (Hz / 64)) << 10) + (lane << 5);
    const char* ldsB  = lds_base + LDSWR_OFF + ((ntl * (Hz / 64)) << 10) + (lane << 5);
    const char* ldsAu = lds_base + LDSU_OFF  + ((mt * (Fz / 64)) << 10) + (lane << 5);
    const char* ldsBi = lds_base + LDSWI_OFF + ((ntl * (Fz / 64)) << 10) + (lane << 5);

    for (int t = 0; t < Tz; ++t) {
        if (t > 0) project_row(h_cur, Wout, bout, out, wave, lane, t - 1);

        for (int e = gtid; e < Bz * Hz; e += NBLK * NTHR) {
            int b = e >> 11, k = e & (Hz - 1);
            float hv = fminf(1.0f, fmaxf(-1.0f, h_cur[e]));
            h_qp[a_pack_addr(b, k, Hz / 64)] = (char)(int)rintf(hv * 7.0f);
        }
        for (int e = gtid; e < Bz * Fz; e += NBLK * NTHR) {
            int b = e >> 7, k = e & (Fz - 1);
            int q = (int)rintf(x[(b * Tz + t) * Fz + k] * 7.0f);
            q = q < -127 ? -127 : (q > 127 ? 127 : q);
            u_qp[a_pack_addr(b, k, Fz / 64)] = (char)q;
        }
        grid_barrier(bar);

        #pragma unroll
        for (int j = 0; j < 16; ++j) {
            int idx = tid + j * NTHR;
            async_b128(LDSA_OFF + idx * 16, h_qp + idx * 16);
        }
        async_b128(LDSU_OFF + tid * 16, u_qp + tid * 16);
        wait_async0();
        __syncthreads();

        v8i acc_res = {0, 0, 0, 0, 0, 0, 0, 0};
        v8i acc_in  = {0, 0, 0, 0, 0, 0, 0, 0};
        for (int kt = 0; kt < Hz / 64; ++kt) {
            v8i a  = *(const v8i*)(ldsA + (kt << 10));
            v8i bm = *(const v8i*)(ldsB + (kt << 10));
            acc_res = __builtin_amdgcn_wmma_i32_16x16x64_iu8(true, a, true, bm,
                                                             acc_res, false, false);
        }
        #pragma unroll
        for (int kt = 0; kt < Fz / 64; ++kt) {
            v8i a  = *(const v8i*)(ldsAu + (kt << 10));
            v8i bm = *(const v8i*)(ldsBi + (kt << 10));
            acc_in = __builtin_amdgcn_wmma_i32_16x16x64_iu8(true, a, true, bm,
                                                            acc_in, false, false);
        }
        #pragma unroll
        for (int r = 0; r < 8; ++r) {
            int b = mt * 16 + half * 8 + r;
            int cr = acc_res[r];
            int ci = acc_in[r];
            cr = cr < -32768 ? -32768 : (cr > 32767 ? 32767 : cr);
            ci = ci < -32768 ? -32768 : (ci > 32767 ? 32767 : ci);
            h_cur[b * Hz + i_col] = tanhf((float)cr * sr + (float)ci * si);
        }
        grid_barrier(bar);
    }
    project_row(h_cur, Wout, bout, out, wave, lane, Tz - 1);
}

extern "C" void kernel_launch(void* const* d_in, const int* in_sizes, int n_in,
                              void* d_out, int out_size, void* d_ws, size_t ws_size,
                              hipStream_t stream) {
    (void)in_sizes; (void)n_in; (void)out_size; (void)ws_size;
    const float* x  = (const float*)d_in[0];
    const int*   Wr = (const int*)d_in[1];
    const int*   Wi = (const int*)d_in[2];
    const float* Sr = (const float*)d_in[3];
    const float* Si = (const float*)d_in[4];
    const float* Ww = (const float*)d_in[5];
    const float* Wb = (const float*)d_in[6];
    float* out = (float*)d_out;

    char* ws = (char*)d_ws;
    char*     Wr_p  = ws + WR_OFF;
    char*     Wi_p  = ws + WI_OFF;
    float*    h_cur = (float*)(ws + HCUR_OFF);
    char*     h_qp  = ws + HQ_OFF;
    char*     u_qp  = ws + UQ_OFF;
    unsigned* bar   = (unsigned*)(ws + BAR_OFF);

    (void)hipFuncSetAttribute((const void*)esn_main_kernel,
                              hipFuncAttributeMaxDynamicSharedMemorySize,
                              (int)LDS_TOTAL);

    pack_res_kernel<<<(Hz * Hz) / NTHR, NTHR, 0, stream>>>(Wr, Wr_p);
    pack_in_kernel<<<(Hz * Fz) / NTHR, NTHR, 0, stream>>>(Wi, Wi_p);
    init_kernel<<<(Bz * Hz + NTHR - 1) / NTHR, NTHR, 0, stream>>>(h_cur, bar);
    esn_main_kernel<<<NBLK, NTHR, LDS_TOTAL, stream>>>(x, Sr, Si, Ww, Wb, Wr_p, Wi_p,
                                                       h_cur, h_qp, u_qp, bar, out);
}
